// TransformerFusionBlock_63118839382086
// MI455X (gfx1250) — matrix operand order repaired and checked by reading, not compiled, not run on hardware
//
#include <hip/hip_runtime.h>
#include <hip/hip_bf16.h>
#include <math.h>

typedef __bf16 bf16;
typedef __attribute__((ext_vector_type(16))) __bf16 v16bf;
typedef __attribute__((ext_vector_type(8)))  float  v8f;

union Frag16 { uint4 u[2]; v16bf v; };

#define D_MODEL 1024
#define NSEQ    256
#define BATCH   64
#define NHEAD   8
#define DKH     128
#define MTOT    (BATCH*NSEQ)   /* 16384 */
#define DFF     (4*D_MODEL)    /* 4096  */

__device__ inline v8f zero8() {
  v8f z = {0.f,0.f,0.f,0.f,0.f,0.f,0.f,0.f};
  return z;
}

// CDNA5 async global->LDS copy (16B per lane), tracked by ASYNCcnt.
__device__ inline void async_b128(unsigned lds_off, const void* gaddr) {
  asm volatile("global_load_async_to_lds_b128 %0, %1, off"
               :: "v"(lds_off), "v"(gaddr)
               : "memory");
}
__device__ inline void wait_async0() {
  asm volatile("s_wait_asynccnt 0" ::: "memory");
}

// ---------------- weight convert + transpose: Wt[n*K+k] = bf16(W[k*N+n]) ------
__global__ __launch_bounds__(256) void k_wconv(const float* __restrict__ W,
                                               bf16* __restrict__ Wt,
                                               int K, int N) {
  long idx = (long)blockIdx.x * 256 + threadIdx.x;
  long total = (long)K * N;
  if (idx >= total) return;
  long n = idx / K, k = idx - n * K;
  Wt[idx] = (bf16)W[k * N + n];   // coalesced write along k
}

// ---------------- LayerNorm fp32 -> bf16 (one row per block, D=1024) ---------
__global__ __launch_bounds__(256) void k_ln(const float* __restrict__ X,
                                            const float* __restrict__ g,
                                            const float* __restrict__ b,
                                            bf16* __restrict__ Y) {
  __shared__ float red[8];
  __shared__ float red2[8];
  long row = blockIdx.x;
  const float* x = X + row * D_MODEL;
  int tid = threadIdx.x, wid = tid >> 5, lane = tid & 31;

  float v[4]; float s = 0.f;
  #pragma unroll
  for (int i = 0; i < 4; ++i) { v[i] = x[tid + 256*i]; s += v[i]; }
  for (int o = 16; o > 0; o >>= 1) s += __shfl_xor(s, o, 32);
  if (lane == 0) red[wid] = s;
  __syncthreads();
  float mean = 0.f;
  #pragma unroll
  for (int i = 0; i < 8; ++i) mean += red[i];
  mean *= (1.f / D_MODEL);

  float vs = 0.f;
  #pragma unroll
  for (int i = 0; i < 4; ++i) { float d = v[i] - mean; vs += d*d; }
  for (int o = 16; o > 0; o >>= 1) vs += __shfl_xor(vs, o, 32);
  if (lane == 0) red2[wid] = vs;
  __syncthreads();
  float var = 0.f;
  #pragma unroll
  for (int i = 0; i < 8; ++i) var += red2[i];
  float rstd = rsqrtf(var * (1.f / D_MODEL) + 1e-5f);

  #pragma unroll
  for (int i = 0; i < 4; ++i) {
    int col = tid + 256*i;
    Y[row * D_MODEL + col] = (bf16)((v[i] - mean) * rstd * g[col] + b[col]);
  }
}

// ---------------- generic bf16 GEMM: C = A[M,K] * Bt[N,K]^T ------------------
// Block tile 256x128, 8 waves in 4x2 grid, wave tile 64x64 (16 WMMA / k-step).
// Double-buffered LDS fed by async global->LDS; one barrier per 32-wide k-step.
// mode 0: outb = bf16(acc + bias)
// mode 1: outb = bf16(gelu(acc + bias))        (exact erf gelu)
// mode 2: outf = coeffs[ia]*res + coeffs[ib]*(acc + bias)   (fp32 out)
__global__ __launch_bounds__(256) void k_gemm(const bf16* __restrict__ A,
                                              const bf16* __restrict__ Bt,
                                              const float* __restrict__ bias,
                                              int M, int N, int K, int mode,
                                              bf16* __restrict__ outb,
                                              float* __restrict__ outf,
                                              const float* __restrict__ res,
                                              const float* __restrict__ coeffs,
                                              int ia, int ib) {
  __shared__ bf16 As[2][256][40];   // 40960 B
  __shared__ bf16 Bs[2][128][40];   // 20480 B  (total 61440 B)
  const int tid  = threadIdx.x;
  const int wid  = tid >> 5, lane = tid & 31;
  const int wm   = wid >> 1, wn = wid & 1;          // 4x2 wave grid
  const long m0  = (long)blockIdx.y * 256;
  const long n0  = (long)blockIdx.x * 128;

  v8f c[4][4];
  #pragma unroll
  for (int i = 0; i < 4; ++i)
    #pragma unroll
    for (int j = 0; j < 4; ++j) c[i][j] = zero8();

  // staging assignment: thread t stages A row t (64B) and half of a B row (32B)
  const int aRow = tid;               // 0..255
  const int bRow = tid >> 1;          // 0..127
  const int bK   = (tid & 1) * 16;    // 0 or 16
  const bf16* gA = A  + (m0 + aRow) * K;
  const bf16* gB = Bt + (n0 + bRow) * K;

  const int arow = lane & 15;
  const int akb  = (lane < 16) ? 0 : 8;
  const int bkb  = (lane < 16) ? 0 : 8;
  const int nK   = K >> 5;

  // prologue: stage k-step 0 into buffer 0
  {
    unsigned la = (unsigned)(uintptr_t)&As[0][aRow][0];
    async_b128(la,      gA);
    async_b128(la + 16, gA + 8);
    async_b128(la + 32, gA + 16);
    async_b128(la + 48, gA + 24);
    unsigned lb = (unsigned)(uintptr_t)&Bs[0][bRow][bK];
    async_b128(lb,      gB + bK);
    async_b128(lb + 16, gB + bK + 8);
  }
  wait_async0();
  __syncthreads();

  for (int kk = 0; kk < nK; ++kk) {
    const int p = kk & 1;
    if (kk + 1 < nK) {                 // stage next k-step into other buffer
      const int k1 = (kk + 1) << 5;
      unsigned la = (unsigned)(uintptr_t)&As[p ^ 1][aRow][0];
      async_b128(la,      gA + k1);
      async_b128(la + 16, gA + k1 + 8);
      async_b128(la + 32, gA + k1 + 16);
      async_b128(la + 48, gA + k1 + 24);
      unsigned lb = (unsigned)(uintptr_t)&Bs[p ^ 1][bRow][bK];
      async_b128(lb,      gB + k1 + bK);
      async_b128(lb + 16, gB + k1 + bK + 8);
    }

    Frag16 a[4], bfr[4];
    #pragma unroll
    for (int mi = 0; mi < 4; ++mi) {
      int r = wm*64 + mi*16 + arow;
      a[mi].u[0] = *reinterpret_cast<const uint4*>(&As[p][r][akb]);
      a[mi].u[1] = *reinterpret_cast<const uint4*>(&As[p][r][akb + 16]);
    }
    #pragma unroll
    for (int ni = 0; ni < 4; ++ni) {
      int r = wn*64 + ni*16 + arow;
      bfr[ni].u[0] = *reinterpret_cast<const uint4*>(&Bs[p][r][bkb]);
      bfr[ni].u[1] = *reinterpret_cast<const uint4*>(&Bs[p][r][bkb + 16]);
    }
    #pragma unroll
    for (int mi = 0; mi < 4; ++mi)
      #pragma unroll
      for (int ni = 0; ni < 4; ++ni)
        c[mi][ni] = __builtin_amdgcn_wmma_f32_16x16x32_bf16(
            false, a[mi].v, false, bfr[ni].v, (short)0, c[mi][ni], false, false);

    wait_async0();        // next buffer fully in LDS (this wave)
    __syncthreads();      // all waves done reading buf p / writing buf p^1
  }

  float ca = 0.f, cb = 0.f;
  if (mode == 2) { ca = coeffs[ia]; cb = coeffs[ib]; }
  const int rowAdd  = (lane < 16) ? 0 : 8;
  const int colLane = lane & 15;
  #pragma unroll
  for (int mi = 0; mi < 4; ++mi)
    #pragma unroll
    for (int ni = 0; ni < 4; ++ni)
      #pragma unroll
      for (int r = 0; r < 8; ++r) {
        long m = m0 + wm*64 + mi*16 + r + rowAdd;
        long n = n0 + wn*64 + ni*16 + colLane;
        float v = c[mi][ni][r] + (bias ? bias[n] : 0.f);
        if (mode == 0) {
          outb[m * N + n] = (bf16)v;
        } else if (mode == 1) {
          float gg = 0.5f * v * (1.f + erff(v * 0.70710678118654752f));
          outb[m * N + n] = (bf16)gg;
        } else {
          outf[m * N + n] = ca * res[m * N + n] + cb * v;
        }
      }
}

// -------- attention scores + softmax: P[bh,q,:] = softmax(scale * Q_bh K_bh^T)
// Q rows private to a wave (global A-fragment loads); all 256 keys staged in LDS.
__global__ __launch_bounds__(256) void k_scores(const bf16* __restrict__ Q,
                                                const bf16* __restrict__ Kb,
                                                bf16* __restrict__ P,
                                                float scale) {
  __shared__ bf16 Ks[256][40];
  int bx = blockIdx.x;
  int rb = bx & 1; int bh = bx >> 1;
  int b  = bh >> 3, h = bh & 7;
  int tid = threadIdx.x, wid = tid >> 5, lane = tid & 31;

  v8f c[16];
  #pragma unroll
  for (int t = 0; t < 16; ++t) c[t] = zero8();

  const long qrow = ((long)(b*NSEQ + rb*128 + wid*16 + (lane & 15))) * D_MODEL + h*DKH;
  const int akb = (lane < 16) ? 0 : 8;
  const int bkb = (lane < 16) ? 0 : 8;

  for (int k0 = 0; k0 < DKH; k0 += 32) {
    {
      const uint4* gk = reinterpret_cast<const uint4*>(
          Kb + ((long)(b*NSEQ + tid)) * D_MODEL + h*DKH + k0);
      uint4 x0 = gk[0], x1 = gk[1], x2 = gk[2], x3 = gk[3];
      uint4* dst = reinterpret_cast<uint4*>(&Ks[tid][0]);
      dst[0] = x0; dst[1] = x1; dst[2] = x2; dst[3] = x3;
    }
    __syncthreads();
    Frag16 a;
    a.u[0] = *reinterpret_cast<const uint4*>(Q + qrow + k0 + akb);
    a.u[1] = *reinterpret_cast<const uint4*>(Q + qrow + k0 + akb + 16);
    #pragma unroll
    for (int t = 0; t < 16; ++t) {
      Frag16 bf;
      int r = t*16 + (lane & 15);
      bf.u[0] = *reinterpret_cast<const uint4*>(&Ks[r][bkb]);
      bf.u[1] = *reinterpret_cast<const uint4*>(&Ks[r][bkb + 16]);
      c[t] = __builtin_amdgcn_wmma_f32_16x16x32_bf16(
          false, a.v, false, bf.v, (short)0, c[t], false, false);
    }
    __syncthreads();
  }

  const int rowAdd = (lane < 16) ? 0 : 8;
  #pragma unroll
  for (int r = 0; r < 8; ++r) {
    float mx = -3.4e38f;
    #pragma unroll
    for (int t = 0; t < 16; ++t) mx = fmaxf(mx, c[t][r]);
    for (int o = 8; o > 0; o >>= 1) mx = fmaxf(mx, __shfl_xor(mx, o, 32));
    float sum = 0.f;
    #pragma unroll
    for (int t = 0; t < 16; ++t) {
      float ev = __expf((c[t][r] - mx) * scale);
      c[t][r] = ev; sum += ev;
    }
    for (int o = 8; o > 0; o >>= 1) sum += __shfl_xor(sum, o, 32);
    float inv = 1.f / sum;
    long q = rb*128 + wid*16 + r + rowAdd;
    long base = ((long)bh * NSEQ + q) * NSEQ;
    #pragma unroll
    for (int t = 0; t < 16; ++t)
      P[base + t*16 + (lane & 15)] = (bf16)(c[t][r] * inv);
  }
}

// -------- O = P * V  (per b,h) with transpose-on-stage for V, merged output --
__global__ __launch_bounds__(256) void k_pv(const bf16* __restrict__ P,
                                            const bf16* __restrict__ V,
                                            bf16* __restrict__ O) {
  __shared__ bf16 Ps[128][40];
  __shared__ bf16 Vt[128][40];
  int bx = blockIdx.x;
  int rb = bx & 1; int bh = bx >> 1;
  int b  = bh >> 3, h = bh & 7;
  int tid = threadIdx.x, wid = tid >> 5, lane = tid & 31;
  int wm = wid >> 1, wn = wid & 1;

  v8f c[2][4];
  #pragma unroll
  for (int i = 0; i < 2; ++i)
    #pragma unroll
    for (int j = 0; j < 4; ++j) c[i][j] = zero8();

  const int ldRow = tid >> 1, ldK = (tid & 1) * 16;
  const int vKey = tid >> 3;          // 0..31
  const int vDk  = (tid & 7) * 16;    // 0..112
  const int arow = lane & 15;
  const int akb  = (lane < 16) ? 0 : 8;
  const int bkb  = (lane < 16) ? 0 : 8;

  for (int k0 = 0; k0 < NSEQ; k0 += 32) {
    {
      const uint4* gp = reinterpret_cast<const uint4*>(
          P + ((long)bh*NSEQ + rb*128 + ldRow) * NSEQ + k0 + ldK);
      *reinterpret_cast<uint4*>(&Ps[ldRow][ldK])     = gp[0];
      *reinterpret_cast<uint4*>(&Ps[ldRow][ldK + 8]) = gp[1];
    }
    {
      Frag16 tmp;
      const uint4* gv = reinterpret_cast<const uint4*>(
          V + ((long)(b*NSEQ) + k0 + vKey) * D_MODEL + h*DKH + vDk);
      tmp.u[0] = gv[0]; tmp.u[1] = gv[1];
      #pragma unroll
      for (int j = 0; j < 16; ++j) Vt[vDk + j][vKey] = tmp.v[j];
    }
    __syncthreads();

    Frag16 a[2], bf[4];
    #pragma unroll
    for (int mi = 0; mi < 2; ++mi) {
      int r = wm*32 + mi*16 + arow;
      a[mi].u[0] = *reinterpret_cast<const uint4*>(&Ps[r][akb]);
      a[mi].u[1] = *reinterpret_cast<const uint4*>(&Ps[r][akb + 16]);
    }
    #pragma unroll
    for (int ni = 0; ni < 4; ++ni) {
      int r = wn*64 + ni*16 + arow;
      bf[ni].u[0] = *reinterpret_cast<const uint4*>(&Vt[r][bkb]);
      bf[ni].u[1] = *reinterpret_cast<const uint4*>(&Vt[r][bkb + 16]);
    }
    #pragma unroll
    for (int mi = 0; mi < 2; ++mi)
      #pragma unroll
      for (int ni = 0; ni < 4; ++ni)
        c[mi][ni] = __builtin_amdgcn_wmma_f32_16x16x32_bf16(
            false, a[mi].v, false, bf[ni].v, (short)0, c[mi][ni], false, false);
    __syncthreads();
  }

  const int rowAdd = (lane < 16) ? 0 : 8;
  const int colLane = lane & 15;
  #pragma unroll
  for (int mi = 0; mi < 2; ++mi)
    #pragma unroll
    for (int ni = 0; ni < 4; ++ni)
      #pragma unroll
      for (int r = 0; r < 8; ++r) {
        int q  = rb*128 + wm*32 + mi*16 + r + rowAdd;
        int dk = wn*64 + ni*16 + colLane;
        O[((long)(b*NSEQ) + q) * D_MODEL + h*DKH + dk] = (bf16)c[mi][ni][r];
      }
}

// -----------------------------------------------------------------------------
extern "C" void kernel_launch(void* const* d_in, const int* in_sizes, int n_in,
                              void* d_out, int out_size, void* d_ws, size_t ws_size,
                              hipStream_t stream) {
  (void)in_sizes; (void)n_in; (void)out_size; (void)ws_size;
  const float* rgb      = (const float*)d_in[0];
  const float* ir       = (const float*)d_in[1];
  const float* ln1_g    = (const float*)d_in[2];
  const float* ln1_b    = (const float*)d_in[3];
  const float* ln2_g    = (const float*)d_in[4];
  const float* ln2_b    = (const float*)d_in[5];
  const float* Wqkv_vis = (const float*)d_in[6];
  const float* bqkv_vis = (const float*)d_in[7];
  const float* Wqkv_ir  = (const float*)d_in[8];
  const float* bqkv_ir  = (const float*)d_in[9];
  const float* Wo_vis   = (const float*)d_in[10];
  const float* bo_vis   = (const float*)d_in[11];
  const float* Wo_ir    = (const float*)d_in[12];
  const float* bo_ir    = (const float*)d_in[13];
  const float* bln_g    = (const float*)d_in[14];
  const float* bln_b    = (const float*)d_in[15];
  const float* W1_vis   = (const float*)d_in[16];
  const float* b1_vis   = (const float*)d_in[17];
  const float* W2_vis   = (const float*)d_in[18];
  const float* b2_vis   = (const float*)d_in[19];
  const float* W1_ir    = (const float*)d_in[20];
  const float* b1_ir    = (const float*)d_in[21];
  const float* W2_ir    = (const float*)d_in[22];
  const float* b2_ir    = (const float*)d_in[23];
  const float* coeffs   = (const float*)d_in[24];
  float* out = (float*)d_out;

  char* w = (char*)d_ws;
  auto alloc = [&](size_t bytes) -> char* {
    char* p = w; w += (bytes + 255) & ~(size_t)255; return p;
  };
  const size_t MD = (size_t)MTOT * D_MODEL;
  const size_t DD = (size_t)D_MODEL * D_MODEL;
  bf16* Xv   = (bf16*)alloc(MD * 2);
  bf16* Xi   = (bf16*)alloc(MD * 2);
  bf16* WqvT = (bf16*)alloc(3 * DD * 2);
  bf16* WqiT = (bf16*)alloc(3 * DD * 2);
  bf16* WovT = (bf16*)alloc(DD * 2);
  bf16* WoiT = (bf16*)alloc(DD * 2);
  bf16* W1vT = (bf16*)alloc((size_t)DFF * D_MODEL * 2);
  bf16* W1iT = (bf16*)alloc((size_t)DFF * D_MODEL * 2);
  bf16* W2vT = (bf16*)alloc((size_t)D_MODEL * DFF * 2);
  bf16* W2iT = (bf16*)alloc((size_t)D_MODEL * DFF * 2);
  bf16* qv = (bf16*)alloc(MD * 2); bf16* kv = (bf16*)alloc(MD * 2); bf16* vv = (bf16*)alloc(MD * 2);
  bf16* qi = (bf16*)alloc(MD * 2); bf16* ki = (bf16*)alloc(MD * 2); bf16* vi = (bf16*)alloc(MD * 2);
  const size_t PE = (size_t)BATCH * NHEAD * NSEQ * NSEQ;
  bf16* Pv = (bf16*)alloc(PE * 2);
  bf16* Pi = (bf16*)alloc(PE * 2);
  bf16* Ov = (bf16*)alloc(MD * 2);
  bf16* Oi = (bf16*)alloc(MD * 2);
  float* attv = (float*)alloc(MD * 4);
  float* atti = (float*)alloc(MD * 4);
  bf16* lav = (bf16*)alloc(MD * 2);
  bf16* lai = (bf16*)alloc(MD * 2);
  bf16* H1v = (bf16*)alloc((size_t)MTOT * DFF * 2);
  bf16* H1i = (bf16*)alloc((size_t)MTOT * DFF * 2);

  auto wc = [&](const float* W, bf16* Wt, int K, int N) {
    long total = (long)K * N;
    int blocks = (int)((total + 255) / 256);
    k_wconv<<<blocks, 256, 0, stream>>>(W, Wt, K, N);
  };
  for (int t = 0; t < 3; ++t) {
    wc(Wqkv_vis + (size_t)t * DD, WqvT + (size_t)t * DD, D_MODEL, D_MODEL);
    wc(Wqkv_ir  + (size_t)t * DD, WqiT + (size_t)t * DD, D_MODEL, D_MODEL);
  }
  wc(Wo_vis, WovT, D_MODEL, D_MODEL);
  wc(Wo_ir,  WoiT, D_MODEL, D_MODEL);
  wc(W1_vis, W1vT, D_MODEL, DFF);
  wc(W1_ir,  W1iT, D_MODEL, DFF);
  wc(W2_vis, W2vT, DFF, D_MODEL);
  wc(W2_ir,  W2iT, DFF, D_MODEL);

  k_ln<<<MTOT, 256, 0, stream>>>(rgb, ln1_g, ln1_b, Xv);
  k_ln<<<MTOT, 256, 0, stream>>>(ir,  ln2_g, ln2_b, Xi);

  auto gemm = [&](const bf16* A, const bf16* Bt, const float* bias,
                  int M, int N, int K, int mode, bf16* ob, float* of,
                  const float* res, int ia, int ib) {
    dim3 g(N / 128, M / 256);
    k_gemm<<<g, 256, 0, stream>>>(A, Bt, bias, M, N, K, mode, ob, of, res, coeffs, ia, ib);
  };

  bf16* qkvv[3] = {qv, kv, vv};
  bf16* qkvi[3] = {qi, ki, vi};
  for (int t = 0; t < 3; ++t) {
    gemm(Xv, WqvT + (size_t)t * DD, bqkv_vis + t * D_MODEL,
         MTOT, D_MODEL, D_MODEL, 0, qkvv[t], nullptr, nullptr, 0, 0);
    gemm(Xi, WqiT + (size_t)t * DD, bqkv_ir + t * D_MODEL,
         MTOT, D_MODEL, D_MODEL, 0, qkvi[t], nullptr, nullptr, 0, 0);
  }

  const float scale = 0.08838834764831845f;   // 1/sqrt(128)
  // att_vis = softmax(q_ir . k_vis),  att_ir = softmax(q_vis . k_ir)
  k_scores<<<BATCH * NHEAD * 2, 256, 0, stream>>>(qi, kv, Pv, scale);
  k_scores<<<BATCH * NHEAD * 2, 256, 0, stream>>>(qv, ki, Pi, scale);
  k_pv<<<BATCH * NHEAD * 2, 256, 0, stream>>>(Pv, vv, Ov);
  k_pv<<<BATCH * NHEAD * 2, 256, 0, stream>>>(Pi, vi, Oi);

  // rgb_att = c0*rgb + c1*(O_vis @ Wo_vis + bo_vis); ir_att likewise
  gemm(Ov, WovT, bo_vis, MTOT, D_MODEL, D_MODEL, 2, nullptr, attv, rgb, 0, 1);
  gemm(Oi, WoiT, bo_ir,  MTOT, D_MODEL, D_MODEL, 2, nullptr, atti, ir,  2, 3);

  k_ln<<<MTOT, 256, 0, stream>>>(attv, bln_g, bln_b, lav);
  k_ln<<<MTOT, 256, 0, stream>>>(atti, bln_g, bln_b, lai);

  gemm(lav, W1vT, b1_vis, MTOT, DFF, D_MODEL, 1, H1v, nullptr, nullptr, 0, 0);
  gemm(lai, W1iT, b1_ir,  MTOT, DFF, D_MODEL, 1, H1i, nullptr, nullptr, 0, 0);

  // rgb_out = c4*rgb_att + c5*(H1 @ W2 + b2) -> d_out[0:MD); ir_out -> d_out[MD:2MD)
  gemm(H1v, W2vT, b2_vis, MTOT, D_MODEL, DFF, 2, nullptr, out,      attv, 4, 5);
  gemm(H1i, W2iT, b2_ir,  MTOT, D_MODEL, DFF, 2, nullptr, out + MD, atti, 6, 7);
}
